// LSTM_21818433864426
// MI455X (gfx1250) — compile-verified, matrix operand order checked by reading, not run on hardware
//
#include <hip/hip_runtime.h>

#define B_   64
#define T_   1024
#define I_   256
#define H_   1024
#define NG_  (4 * H_)      // 4096 gate columns (f,i,g,o)
#define KT_  (I_ + H_)     // 1280 total reduction dim
#define WGS_ 64            // persistent workgroups (one per 16 h-columns)
#define TPB_ 256           // 8 waves of 32
#define LDK_ (H_ + 8)      // padded LDS row stride (bf16 elems) -> conflict-free ds_load_b128

typedef __attribute__((ext_vector_type(16))) __bf16   v16bf;
typedef __attribute__((ext_vector_type(8)))  float    v8f;
typedef __attribute__((ext_vector_type(4)))  unsigned u32x4;

union FragU { u32x4 u[2]; v16bf v; };

// Load a 16x32 bf16 WMMA fragment (A or B, mirror layouts) from a row-major
// matrix. Per ISA 7.12.2: lane L holds row/col (L&15); K = kbase + (e<8?e:e+8)
// with kbase = k0 + (L<16 ? 0 : 8). Two 16B vector loads per lane.
__device__ __forceinline__ v16bf load_frag(const unsigned short* row, int k) {
  FragU f;
  f.u[0] = *reinterpret_cast<const u32x4*>(row + k);
  f.u[1] = *reinterpret_cast<const u32x4*>(row + k + 16);
  return f.v;
}

__device__ __forceinline__ unsigned short f2bf(float x) {
  union { float f; unsigned u; } c; c.f = x;
  unsigned r = c.u + 0x7FFFu + ((c.u >> 16) & 1u);   // round-to-nearest-even
  return (unsigned short)(r >> 16);
}

__device__ __forceinline__ float sigm(float x) { return 1.0f / (1.0f + __expf(-x)); }

// ---------------------------------------------------------------------------
// Zero h0 double-buffer + grid barrier counter (must run every launch).
__global__ void init_kernel(unsigned short* __restrict__ hbf, unsigned* __restrict__ cnt) {
  int i = blockIdx.x * TPB_ + threadIdx.x;
  if (i < 2 * B_ * H_) hbf[i] = 0;
  if (i == 0) *cnt = 0u;
}

// ---------------------------------------------------------------------------
// Pack transposed, gate-concatenated weights: Wt[n][k], n in [0,4096), k in
// [0,1280). k<256 -> Wx-gate row k, else Wh-gate row k-256. bf16.
__global__ void convert_w_kernel(const float* __restrict__ Wfx, const float* __restrict__ Wfh,
                                 const float* __restrict__ Wix, const float* __restrict__ Wih,
                                 const float* __restrict__ Wgx, const float* __restrict__ Wgh,
                                 const float* __restrict__ Wox, const float* __restrict__ Woh,
                                 unsigned short* __restrict__ Wt) {
  size_t i = (size_t)blockIdx.x * TPB_ + threadIdx.x;
  if (i >= (size_t)NG_ * KT_) return;
  int n = (int)(i / KT_);
  int k = (int)(i % KT_);
  int g = n >> 10;
  int col = n & (H_ - 1);
  const float* Wx; const float* Wh;
  switch (g) {
    case 0:  Wx = Wfx; Wh = Wfh; break;
    case 1:  Wx = Wix; Wh = Wih; break;
    case 2:  Wx = Wgx; Wh = Wgh; break;
    default: Wx = Wox; Wh = Woh; break;
  }
  float v = (k < I_) ? Wx[(size_t)k * H_ + col] : Wh[(size_t)(k - I_) * H_ + col];
  Wt[i] = f2bf(v);
}

__global__ void convert_x_kernel(const float* __restrict__ x, unsigned short* __restrict__ xbf) {
  size_t i = (size_t)blockIdx.x * TPB_ + threadIdx.x;
  if (i < (size_t)B_ * T_ * I_) xbf[i] = f2bf(x[i]);
}

// ---------------------------------------------------------------------------
// Persistent LSTM kernel. WG w owns h columns [w*16, w*16+16) -> gate tiles at
// n = g*1024 + w*16 for g=0..3, all 4 M-tiles (B=64). 16 tiles of 16x16, 8
// waves, 2 tiles/wave (shared A fragment). Wx slice register-resident, Wh
// slice LDS-resident. Split grid barrier: per-step wait is issued only before
// the h-dependent phase, so the x-projection WMMAs overlap barrier latency.
__global__ __launch_bounds__(TPB_, 1)
void lstm_persistent(const unsigned short* __restrict__ xbf,   // [B][T][I] bf16
                     const unsigned short* __restrict__ Wt,    // [4096][1280] bf16
                     unsigned short* __restrict__ hbf,         // [2][B][H] bf16
                     const float* __restrict__ pbf, const float* __restrict__ pbi,
                     const float* __restrict__ pbg, const float* __restrict__ pbo,
                     float* __restrict__ hlast,                // [H] fp32, batch row 63 @ t=T-1
                     unsigned* __restrict__ barrier_cnt) {
  __shared__ unsigned short Wl[64 * LDK_];   // 132,096 B : Wh slice, 64 gate-cols x 1024 K
  __shared__ float gsh[4][4][256];           //  16,384 B : [gate][mtile][row*16+col]

  const int wg   = blockIdx.x;
  const int tid  = threadIdx.x;
  const int wave = tid >> 5;
  const int lane = tid & 31;

  // --- stage Wh slice (K = 256..1279 of the 64 gate columns) into LDS -------
  for (int i = tid; i < 64 * (H_ / 8); i += TPB_) {
    int nl = i >> 7;              // local gate-column 0..63
    int kk = (i & 127) << 3;      // K offset 0..1016 step 8
    int g = nl >> 4, col = nl & 15;
    int n = (g << 10) + (wg << 4) + col;
    *reinterpret_cast<u32x4*>(&Wl[nl * LDK_ + kk]) =
        *reinterpret_cast<const u32x4*>(&Wt[(size_t)n * KT_ + I_ + kk]);
  }
  __syncthreads();

  // --- per-wave tile assignment --------------------------------------------
  const int mt   = wave >> 1;            // M tile 0..3 (batch rows mt*16..)
  const int g0   = (wave & 1) << 1;      // gate pair {0,1} or {2,3}
  const int g1   = g0 + 1;
  const int half = lane >> 4;            // 0 | 1
  const int koff = half << 3;            // fragment kbase offset per ISA layout
  const int col16 = lane & 15;
  const int aRow  = (mt << 4) + col16;   // batch row this lane feeds (A matrix)

  const unsigned short* bldsg0 = &Wl[((g0 << 4) + col16) * LDK_];
  const unsigned short* bldsg1 = &Wl[((g1 << 4) + col16) * LDK_];
  const unsigned short* bx0 = Wt + (size_t)((g0 << 10) + (wg << 4) + col16) * KT_;
  const unsigned short* bx1 = Wt + (size_t)((g1 << 10) + (wg << 4) + col16) * KT_;
  const unsigned short* xrow_base = xbf + (size_t)aRow * (T_ * I_);
  const unsigned short* hrow0 = hbf + aRow * H_;
  const unsigned short* hrow1 = hbf + B_ * H_ + aRow * H_;

  // --- x-projection weights register-resident for the whole sequence -------
  // 16 fragments x 8 VGPRs = 128 VGPRs; loop-invariant over all 1024 steps.
  v16bf wxr0[8], wxr1[8];
#pragma unroll
  for (int kc = 0; kc < I_ / 32; ++kc) {
    wxr0[kc] = load_frag(bx0, kc * 32 + koff);
    wxr1[kc] = load_frag(bx1, kc * 32 + koff);
  }

  // --- per-thread elementwise mapping (4 (b,j) cells, constant over t) ------
  float Creg[4] = {0.f, 0.f, 0.f, 0.f};
  float bfr[4], bir[4], bgr[4], bor[4];
  int brow_[4], emt_[4], li_[4], jg_[4];
#pragma unroll
  for (int q = 0; q < 4; ++q) {
    int e = tid + (q << 8);
    int brow = e >> 4, c = e & 15;
    brow_[q] = brow;
    emt_[q]  = brow >> 4;
    li_[q]   = ((brow & 15) << 4) + c;
    int jg   = (wg << 4) + c;
    jg_[q]   = jg;
    bfr[q] = pbf[jg]; bir[q] = pbi[jg]; bgr[q] = pbg[jg]; bor[q] = pbo[jg];
  }

  for (int t = 0; t < T_; ++t) {
    const unsigned short* xr = xrow_base + (size_t)t * I_;
    __builtin_prefetch(xr + I_, 0, 1);   // global_prefetch next timestep's x row

    // ---- x projection: K = 0..255, A from x, B from registers -------------
    // Independent of other WGs' h writes -> runs before the barrier wait and
    // hides the barrier's L2 round-trip latency.
    v8f acc0 = {};
    v8f acc1 = {};
#pragma unroll
    for (int kc = 0; kc < I_ / 32; ++kc) {
      v16bf av = load_frag(xr, kc * 32 + koff);
      acc0 = __builtin_amdgcn_wmma_f32_16x16x32_bf16(false, av, false, wxr0[kc], (short)0, acc0, false, false);
      acc1 = __builtin_amdgcn_wmma_f32_16x16x32_bf16(false, av, false, wxr1[kc], (short)0, acc1, false, false);
    }

    // ---- split-barrier WAIT: h[t] must be complete from all WGs -----------
    if (t > 0) {
      unsigned waitTgt = (unsigned)t * WGS_;
      if (tid == 0) {
        while (__hip_atomic_load(barrier_cnt, __ATOMIC_ACQUIRE, __HIP_MEMORY_SCOPE_AGENT) < waitTgt)
          __builtin_amdgcn_s_sleep(1);
      }
      __syncthreads();
      __threadfence();   // acquire per wave: invalidate stale h lines in vL0
    }

    // ---- recurrent part: K = 256..1279, A from h_prev, B from LDS ---------
    const unsigned short* hr = (t & 1) ? hrow1 : hrow0;
#pragma unroll 4
    for (int k = 0; k < H_; k += 32) {
      v16bf av = load_frag(hr, k + koff);
      v16bf b0 = load_frag(bldsg0, k + koff);
      v16bf b1 = load_frag(bldsg1, k + koff);
      acc0 = __builtin_amdgcn_wmma_f32_16x16x32_bf16(false, av, false, b0, (short)0, acc0, false, false);
      acc1 = __builtin_amdgcn_wmma_f32_16x16x32_bf16(false, av, false, b1, (short)0, acc1, false, false);
    }

    // ---- stage accumulators to LDS (D layout: VGPR r, M = r + 8*half) -----
#pragma unroll
    for (int r = 0; r < 8; ++r) {
      int row = r + (half << 3);
      gsh[g0][mt][(row << 4) + col16] = acc0[r];
      gsh[g1][mt][(row << 4) + col16] = acc1[r];
    }
    __syncthreads();

    // ---- elementwise LSTM cell update (gates all sigmoid, per reference) --
    unsigned short* hn = hbf + ((t ^ 1) & 1) * (B_ * H_);
#pragma unroll
    for (int q = 0; q < 4; ++q) {
      float f  = sigm(gsh[0][emt_[q]][li_[q]] + bfr[q]);
      float iv = sigm(gsh[1][emt_[q]][li_[q]] + bir[q]);
      float gv = sigm(gsh[2][emt_[q]][li_[q]] + bgr[q]);
      float ov = sigm(gsh[3][emt_[q]][li_[q]] + bor[q]);
      float Cn = gv * iv + Creg[q] * f;
      Creg[q] = Cn;
      float hv = ov * tanhf(Cn);
      hn[brow_[q] * H_ + jg_[q]] = f2bf(hv);
      if (t == (T_ - 1) && brow_[q] == (B_ - 1)) hlast[jg_[q]] = hv;
    }

    // ---- split-barrier SIGNAL: h[t+1] published ---------------------------
    __threadfence();   // release per wave: flush h stores toward L2
    __syncthreads();   // also separates gsh reads from next step's writes
    if (tid == 0)
      __hip_atomic_fetch_add(barrier_cnt, 1u, __ATOMIC_ACQ_REL, __HIP_MEMORY_SCOPE_AGENT);
  }
}

// ---------------------------------------------------------------------------
// y[p] = sum_h hlast[h] * Wph[h][p] + bp[p]   (one block per class, p<10)
__global__ void head_kernel(const float* __restrict__ hlast, const float* __restrict__ Wph,
                            const float* __restrict__ bp, float* __restrict__ y) {
  __shared__ float red[TPB_];
  int tid = threadIdx.x;
  int p = blockIdx.x;
  float s = 0.f;
  for (int h = tid; h < H_; h += TPB_) s += hlast[h] * Wph[(size_t)h * 10 + p];
  red[tid] = s;
  __syncthreads();
  for (int off = TPB_ / 2; off > 0; off >>= 1) {
    if (tid < off) red[tid] += red[tid + off];
    __syncthreads();
  }
  if (tid == 0) y[p] = red[0] + bp[p];
}

// ---------------------------------------------------------------------------
extern "C" void kernel_launch(void* const* d_in, const int* in_sizes, int n_in,
                              void* d_out, int out_size, void* d_ws, size_t ws_size,
                              hipStream_t stream) {
  (void)in_sizes; (void)n_in; (void)out_size; (void)ws_size;
  const float* x   = (const float*)d_in[0];
  const float* Wfx = (const float*)d_in[1];
  const float* Wfh = (const float*)d_in[2];
  const float* bf  = (const float*)d_in[3];
  const float* Wix = (const float*)d_in[4];
  const float* Wih = (const float*)d_in[5];
  const float* bi  = (const float*)d_in[6];
  const float* Wgx = (const float*)d_in[7];
  const float* Wgh = (const float*)d_in[8];
  const float* bg  = (const float*)d_in[9];
  const float* Wox = (const float*)d_in[10];
  const float* Woh = (const float*)d_in[11];
  const float* bo  = (const float*)d_in[12];
  const float* Wph = (const float*)d_in[13];
  const float* bp  = (const float*)d_in[14];
  float* y = (float*)d_out;

  // workspace layout (all regions 256B-aligned)
  char* ws = (char*)d_ws;
  size_t offWt = 0;
  size_t offX  = offWt + (size_t)NG_ * KT_ * 2;       // 10,485,760
  size_t offH  = offX  + (size_t)B_ * T_ * I_ * 2;    // +33,554,432
  size_t offHL = offH  + (size_t)2 * B_ * H_ * 2;     // +262,144
  size_t offC  = offHL + 4096;                        // counter
  unsigned short* Wt  = (unsigned short*)(ws + offWt);
  unsigned short* xbf = (unsigned short*)(ws + offX);
  unsigned short* hbf = (unsigned short*)(ws + offH);
  float*    hlast = (float*)(ws + offHL);
  unsigned* cnt   = (unsigned*)(ws + offC);

  init_kernel<<<(2 * B_ * H_ + TPB_ - 1) / TPB_, TPB_, 0, stream>>>(hbf, cnt);
  convert_w_kernel<<<(unsigned)(((size_t)NG_ * KT_ + TPB_ - 1) / TPB_), TPB_, 0, stream>>>(
      Wfx, Wfh, Wix, Wih, Wgx, Wgh, Wox, Woh, Wt);
  convert_x_kernel<<<(unsigned)(((size_t)B_ * T_ * I_ + TPB_ - 1) / TPB_), TPB_, 0, stream>>>(x, xbf);
  lstm_persistent<<<WGS_, TPB_, 0, stream>>>(xbf, Wt, hbf, bf, bi, bg, bo, hlast, cnt);
  head_kernel<<<10, TPB_, 0, stream>>>(hlast, Wph, bp, y);
}
